// TransformerBlock_87222195847552
// MI455X (gfx1250) — matrix operand order repaired and checked by reading, not compiled, not run on hardware
//
#include <hip/hip_runtime.h>
#include <hip/hip_bf16.h>
#include <math.h>

typedef __attribute__((ext_vector_type(16))) _Float16 v16h;
typedef __attribute__((ext_vector_type(8)))  _Float16 v8h;
typedef __attribute__((ext_vector_type(8)))  float    v8f;
typedef __attribute__((address_space(3)))    _Float16 lds_f16;

#define DIM   512
#define SEQ   2048
#define BATCH 4
#define TOK   (BATCH*SEQ)      // 8192
#define HEADS 8
#define HD    64
#define TRIPLE (3*DIM)         // 1536
#define FF    (2*DIM)          // 1024
#define QK_LD 1024             // q|k token-major buffer width

// ---------------------------------------------------------------------------
// CDNA5 async copy: global -> LDS, 16 bytes per lane, tracked by ASYNCcnt.
// ---------------------------------------------------------------------------
__device__ __forceinline__ void async_cp16(const _Float16* g, _Float16* l) {
  lds_f16* ll = (lds_f16*)l;    // addrspacecast generic -> LDS (32-bit)
  asm volatile("global_load_async_to_lds_b128 %0, %1, off"
               :: "v"(ll), "v"(g) : "memory");
}
__device__ __forceinline__ void wait_async0() {
  asm volatile("s_wait_asynccnt 0x0" ::: "memory");
}

// raw max without NaN canonicalization (saves a v_max per operand)
__device__ __forceinline__ float maxf_raw(float a, float b) {
  float r;
  asm("v_max_num_f32 %0, %1, %2" : "=v"(r) : "v"(a), "v"(b));
  return r;
}

// ---------------------------------------------------------------------------
// WMMA helper: D = A(16x32 f16) * B(32x16 f16) + C(16x16 f32)
// ---------------------------------------------------------------------------
__device__ __forceinline__ v8f wmma16(v16h a, v16h b, v8f c) {
  return __builtin_amdgcn_wmma_f32_16x16x32_f16(
      false, a, false, b, (short)0, c, false, false);
}

// ---------------------------------------------------------------------------
// DPP 16-lane butterfly reductions (stay within each 16-lane C-row group).
// xor1=quad_perm(1,0,3,2)=0xB1 ; xor2=quad_perm(2,3,0,1)=0x4E ;
// xor7=row_half_mirror=0x141   ; xor15=row_mirror=0x140
// ---------------------------------------------------------------------------
template <int CTRL>
__device__ __forceinline__ float dpp_mov(float v) {
  return __builtin_bit_cast(float,
      __builtin_amdgcn_mov_dpp(__builtin_bit_cast(int, v), CTRL, 0xF, 0xF, true));
}
__device__ __forceinline__ float redmax16(float v) {
  v = maxf_raw(v, dpp_mov<0xB1>(v));
  v = maxf_raw(v, dpp_mov<0x4E>(v));
  v = maxf_raw(v, dpp_mov<0x141>(v));
  v = maxf_raw(v, dpp_mov<0x140>(v));
  return v;
}
__device__ __forceinline__ float redsum16(float v) {
  v += dpp_mov<0xB1>(v);
  v += dpp_mov<0x4E>(v);
  v += dpp_mov<0x141>(v);
  v += dpp_mov<0x140>(v);
  return v;
}

// A fragment (16x32 row-major, leading dim ld). lane l: row=l%16,
// K(e) = e + (l<16 ? 0 : 16)  -> 16 contiguous elements.
__device__ __forceinline__ v16h frag_A(const _Float16* p, int ld) {
  int lane = threadIdx.x & 31;
  const _Float16* q = p + (lane & 15) * ld + ((lane & 16) ? 16 : 0);
  v16h a;
#pragma unroll
  for (int e = 0; e < 8; ++e) { a[e] = q[e]; a[e + 8] = q[e + 8]; }
  return a;
}

// B fragment from a row-major [n][k] source (i.e. B = M^T), leading dim ld.
// lane l: col(n) = l%16 ; K(e) = e + (l<16?0:16) -> 16 contiguous elements.
__device__ __forceinline__ v16h frag_Bt(const _Float16* p, int ld) {
  int lane = threadIdx.x & 31;
  const _Float16* q = p + (lane & 15) * ld + ((lane & 16) ? 16 : 0);
  v16h b;
#pragma unroll
  for (int e = 0; e < 16; ++e) b[e] = q[e];
  return b;
}

// ---------------------------------------------------------------------------
// fp32 [K][N] -> f16 transposed [N][K]  (weights; LDS-tiled transpose)
// ---------------------------------------------------------------------------
__global__ __launch_bounds__(256) void transpose_cvt(const float* __restrict__ w,
                                                     _Float16* __restrict__ wt,
                                                     int K, int N) {
  __shared__ float t[32][33];
  int k0 = blockIdx.y * 32, n0 = blockIdx.x * 32;
  int tx = threadIdx.x & 31, ty = threadIdx.x >> 5;  // ty 0..7
#pragma unroll
  for (int i = 0; i < 32; i += 8)
    t[ty + i][tx] = w[(size_t)(k0 + ty + i) * N + n0 + tx];
  __syncthreads();
#pragma unroll
  for (int i = 0; i < 32; i += 8)
    wt[(size_t)(n0 + ty + i) * K + k0 + tx] = (_Float16)t[tx][ty + i];
}

// ---------------------------------------------------------------------------
// LayerNorm over last dim (512) fp32 -> f16. One wave per row.
// ---------------------------------------------------------------------------
__global__ __launch_bounds__(256) void ln_f16(const float* __restrict__ x,
                                              const float* __restrict__ g,
                                              const float* __restrict__ b,
                                              _Float16* __restrict__ o) {
  int row  = blockIdx.x * 8 + (threadIdx.x >> 5);
  int lane = threadIdx.x & 31;
  const float* xr = x + (size_t)row * DIM;
  float v[16], s = 0.f, ss = 0.f;
#pragma unroll
  for (int e = 0; e < 16; ++e) {
    float t = xr[lane * 16 + e];
    v[e] = t; s += t; ss += t * t;
  }
#pragma unroll
  for (int m = 1; m < 32; m <<= 1) { s += __shfl_xor(s, m); ss += __shfl_xor(ss, m); }
  float mu  = s * (1.0f / DIM);
  float var = ss * (1.0f / DIM) - mu * mu;
  float rs  = rsqrtf(var + 1e-5f);
  _Float16* orow = o + (size_t)row * DIM;
#pragma unroll
  for (int e = 0; e < 16; ++e) {
    int c = lane * 16 + e;
    orow[c] = (_Float16)((v[e] - mu) * rs * g[c] + b[c]);
  }
}

// ---------------------------------------------------------------------------
// Tiled WMMA GEMM: C[M,N] = epi(A[M,K] @ BT[N,K]^T).
// Block = 256 threads (8 waves), 256x128 block tile, wave = 64x64 (4x4 frag
// tiles -> 16 WMMA per 16 ds_load_b128 per K-step: 1.5x the FLOPs per LDS byte
// of a 32x64 wave tile). Async double-buffered LDS staging
// (global_load_async_to_lds_b128), one barrier per K-step.
// EPI: 1 -> f32 out (+bias+residual)
//      2 -> f16 out (+bias+exact GELU)
//      3 -> QKV split: cols <1024 token-major qk f16; cols >=1024 transposed vT
// ---------------------------------------------------------------------------
#define LD_S 40   // padded LDS leading dim (f16), 80B rows keep 16B alignment
template <int EPI>
__global__ __launch_bounds__(256) void gemm_wmma(
    const _Float16* __restrict__ A,   // [M][K]
    const _Float16* __restrict__ BT,  // [N][K]
    const float* __restrict__ bias, const float* __restrict__ res,
    float* __restrict__ outF, _Float16* __restrict__ outH,
    _Float16* __restrict__ outV,
    int M, int N, int K) {
  __shared__ _Float16 As[2][256 * LD_S];
  __shared__ _Float16 Bs[2][128 * LD_S];

  int tid = threadIdx.x, lane = tid & 31, wid = tid >> 5;
  int wm = wid >> 1;                 // 0..3  (M strip of 64)
  int wn = wid & 1;                  // 0..1  (N strip of 64)
  int m0 = blockIdx.y * 256;
  int n0 = blockIdx.x * 128;

  // staging: A -> one row (32 f16 = 4 b128) per thread; B -> half row per thread
  int br = tid >> 1, bc = (tid & 1) * 16;
  const _Float16* pa = A + (size_t)(m0 + tid) * K;
  const _Float16* pb = BT + (size_t)(n0 + br) * K + bc;

  // issue tile 0 async copies
#pragma unroll
  for (int c = 0; c < 4; ++c) async_cp16(pa + c * 8, &As[0][tid * LD_S + c * 8]);
  async_cp16(pb,     &Bs[0][br * LD_S + bc]);
  async_cp16(pb + 8, &Bs[0][br * LD_S + bc + 8]);

  v8f acc[4][4];
#pragma unroll
  for (int i = 0; i < 4; ++i)
#pragma unroll
    for (int j = 0; j < 4; ++j) acc[i][j] = (v8f)0.0f;

  int buf = 0;
  for (int kt = 0;;) {
    wait_async0();                         // my tile-t copies landed in LDS
    __syncthreads();                       // block-wide: tile t visible
    bool more = (kt + 32) < K;
    if (more) {                            // issue tile t+1 into other buffer
      const _Float16* na = pa + kt + 32;
      const _Float16* nb = pb + kt + 32;
#pragma unroll
      for (int c = 0; c < 4; ++c)
        async_cp16(na + c * 8, &As[buf ^ 1][tid * LD_S + c * 8]);
      async_cp16(nb,     &Bs[buf ^ 1][br * LD_S + bc]);
      async_cp16(nb + 8, &Bs[buf ^ 1][br * LD_S + bc + 8]);
      if (kt + 64 < K) {                   // distance-2 prefetch
        __builtin_prefetch(pa + kt + 64, 0, 3);
        __builtin_prefetch(pb + kt + 64, 0, 3);
      }
    }

    v16h af[4], bf[4];
#pragma unroll
    for (int mi = 0; mi < 4; ++mi)
      af[mi] = frag_A(&As[buf][(wm * 64 + mi * 16) * LD_S], LD_S);
#pragma unroll
    for (int n = 0; n < 4; ++n)
      bf[n] = frag_Bt(&Bs[buf][(wn * 64 + n * 16) * LD_S], LD_S);
#pragma unroll
    for (int mi = 0; mi < 4; ++mi)
#pragma unroll
      for (int n = 0; n < 4; ++n)
        acc[mi][n] = wmma16(af[mi], bf[n], acc[mi][n]);

    kt += 32; buf ^= 1;
    if (!more) break;
  }

  int lc = lane & 15;
  int lr = (lane & 16) ? 8 : 0;
#pragma unroll
  for (int mi = 0; mi < 4; ++mi)
#pragma unroll
    for (int n = 0; n < 4; ++n) {
      int colb = n0 + wn * 64 + n * 16;         // tile column base
      int rowb = m0 + wm * 64 + mi * 16 + lr;   // this lane's first row
      if (EPI == 1) {
#pragma unroll
        for (int r = 0; r < 8; ++r) {
          size_t idx = (size_t)(rowb + r) * N + colb + lc;
          outF[idx] = acc[mi][n][r] + bias[colb + lc] + res[idx];
        }
      } else if (EPI == 2) {
#pragma unroll
        for (int r = 0; r < 8; ++r) {
          float gv = acc[mi][n][r] + bias[colb + lc];
          outH[(size_t)(rowb + r) * N + colb + lc] =
              (_Float16)(0.5f * gv * (1.0f + erff(gv * 0.70710678118654752f)));
        }
      } else {  // EPI == 3 : QKV split
        if (colb < 2 * DIM) {  // Q|K -> token-major [TOK][1024]
#pragma unroll
          for (int r = 0; r < 8; ++r)
            outH[(size_t)(rowb + r) * QK_LD + colb + lc] = (_Float16)acc[mi][n][r];
        } else {               // V -> vT[(b*8+h)*64+d][n] ; 8 tokens contiguous
          v8h pk;
#pragma unroll
          for (int r = 0; r < 8; ++r) pk[r] = (_Float16)acc[mi][n][r];
          int bidx = rowb >> 11;          // batch
          int nin  = rowb & (SEQ - 1);    // position within sequence
          size_t vrow = (size_t)(bidx * DIM + (colb + lc - 2 * DIM));
          *(v8h*)&outV[vrow * SEQ + nin] = pk;
        }
      }
    }
}

// ---------------------------------------------------------------------------
// Flash attention. qk: [8192][1024] f16 (Q|K, cols h*64+d / 512+h*64+d).
// vT: [32 bh][64 d][2048 n] f16.  Block: one (b,h) x 128 queries (8 waves x16).
// 1/sqrt(64) folded into Q. DPP softmax reductions. Async double-buffered
// K/V staging. attO: [8192][512] f16 token-major.
// ---------------------------------------------------------------------------
#define KV_LD 72   // 144B rows: 16B-aligned b128 granules
#define VT_LD 40   // 80B rows
#define P_LD  36
__global__ __launch_bounds__(256) void attn_wmma(const _Float16* __restrict__ qk,
                                                 const _Float16* __restrict__ vT,
                                                 _Float16* __restrict__ attO) {
  __shared__ _Float16 Kt[2][32 * KV_LD];     // [key][dim]
  __shared__ _Float16 VTt[2][64 * VT_LD];    // [dim][key]
  __shared__ _Float16 Pscr[8 * 16 * P_LD];   // per-wave P relayout scratch

  int tid = threadIdx.x, lane = tid & 31, wid = tid >> 5;
  int bh = blockIdx.y;
  int b = bh >> 3, h = bh & 7;
  int tb = b * SEQ;
  int q0 = blockIdx.x * 128 + wid * 16;

  const _Float16* qbase = qk + (size_t)(tb + q0) * QK_LD + h * HD;
  v16h qa0 = frag_A(qbase, QK_LD);
  v16h qa1 = frag_A(qbase + 32, QK_LD);
  {  // fold the 1/sqrt(HD) score scale into Q once (packed f16 muls)
    const _Float16 s8 = (_Float16)0.125f;
#pragma unroll
    for (int e = 0; e < 16; ++e) { qa0[e] *= s8; qa1[e] *= s8; }
  }

  _Float16* ps = &Pscr[wid * 16 * P_LD];

  v8f acc[4];
#pragma unroll
  for (int n = 0; n < 4; ++n) acc[n] = (v8f)0.0f;
  float mrow[8], lrow[8];
#pragma unroll
  for (int r = 0; r < 8; ++r) { mrow[r] = -1e30f; lrow[r] = 0.f; }

  int lc = lane & 15;
  int half8 = (lane & 16) ? 8 : 0;

  // staging coords (one b128 per thread per tile)
  int krow = tid >> 3, kc0 = (tid & 7) * 8;   // K tile: 32 x 64
  int vrow = tid >> 2, vc0 = (tid & 3) * 8;   // VT tile: 64 x 32
  const _Float16* kp = qk + (size_t)(tb + krow) * QK_LD + DIM + h * HD + kc0;
  const _Float16* vp = vT + (size_t)(bh * HD + vrow) * SEQ + vc0;

  async_cp16(kp, &Kt[0][krow * KV_LD + kc0]);
  async_cp16(vp, &VTt[0][vrow * VT_LD + vc0]);

  int buf = 0;
  for (int j = 0;;) {
    wait_async0();
    __syncthreads();                      // staged K/V tile visible
    bool more = (j + 32) < SEQ;
    if (more) {                           // issue next tile into other buffer
      async_cp16(kp + (size_t)(j + 32) * QK_LD, &Kt[buf ^ 1][krow * KV_LD + kc0]);
      async_cp16(vp + (j + 32),                 &VTt[buf ^ 1][vrow * VT_LD + vc0]);
      if (j + 64 < SEQ) {
        __builtin_prefetch(kp + (size_t)(j + 64) * QK_LD, 0, 3);
        __builtin_prefetch(vp + j + 64, 0, 3);
      }
    }

    // S = Q @ K^T for 32 keys (two 16x16 tiles); contiguous frag loads
    v8f s0 = (v8f)0.0f, s1 = (v8f)0.0f;
    s0 = wmma16(qa0, frag_Bt(&Kt[buf][0 * KV_LD + 0], KV_LD), s0);
    s0 = wmma16(qa1, frag_Bt(&Kt[buf][0 * KV_LD + 32], KV_LD), s0);
    s1 = wmma16(qa0, frag_Bt(&Kt[buf][16 * KV_LD + 0], KV_LD), s1);
    s1 = wmma16(qa1, frag_Bt(&Kt[buf][16 * KV_LD + 32], KV_LD), s1);

    // online softmax; row reductions via DPP butterflies (VALU only)
#pragma unroll
    for (int r = 0; r < 8; ++r) {
      float v0 = s0[r];
      float v1 = s1[r];
      float mx = redmax16(maxf_raw(v0, v1));
      float mn   = maxf_raw(mrow[r], mx);
      float corr = __expf(mrow[r] - mn);
      float p0   = __expf(v0 - mn);
      float p1   = __expf(v1 - mn);
      float rs   = redsum16(p0 + p1);
      lrow[r] = lrow[r] * corr + rs;
      mrow[r] = mn;
#pragma unroll
      for (int n = 0; n < 4; ++n) acc[n][r] *= corr;
      int prow = r + half8;
      ps[prow * P_LD + lc]      = (_Float16)p0;   // keys j..j+15
      ps[prow * P_LD + lc + 16] = (_Float16)p1;   // keys j+16..j+31
    }

    // O += P(16x32) @ V(32x64); V frags contiguous from VTt
    v16h pa = frag_A(ps, P_LD);
#pragma unroll
    for (int n = 0; n < 4; ++n)
      acc[n] = wmma16(pa, frag_Bt(&VTt[buf][(n * 16) * VT_LD], VT_LD), acc[n]);

    j += 32; buf ^= 1;
    if (!more) break;
  }

  // normalize + write attended (token-major, column h*64+d)
#pragma unroll
  for (int n = 0; n < 4; ++n) {
    int col = h * HD + n * 16 + lc;
#pragma unroll
    for (int r = 0; r < 8; ++r) {
      int token = tb + q0 + r + half8;
      attO[(size_t)token * DIM + col] = (_Float16)(acc[n][r] / lrow[r]);
    }
  }
}

// ---------------------------------------------------------------------------
// Host launch
// ---------------------------------------------------------------------------
extern "C" void kernel_launch(void* const* d_in, const int* in_sizes, int n_in,
                              void* d_out, int out_size, void* d_ws, size_t ws_size,
                              hipStream_t stream) {
  (void)in_sizes; (void)n_in; (void)out_size; (void)ws_size;
  const float* x     = (const float*)d_in[0];
  const float* ln1_g = (const float*)d_in[1];
  const float* ln1_b = (const float*)d_in[2];
  const float* w_qkv = (const float*)d_in[3];
  const float* w_out = (const float*)d_in[4];
  const float* b_out = (const float*)d_in[5];
  const float* ln2_g = (const float*)d_in[6];
  const float* ln2_b = (const float*)d_in[7];
  const float* w1    = (const float*)d_in[8];
  const float* b1    = (const float*)d_in[9];
  const float* w2    = (const float*)d_in[10];
  const float* b2    = (const float*)d_in[11];
  float* out = (float*)d_out;
  char* ws = (char*)d_ws;

  // workspace layout (256B-aligned offsets)
  _Float16* wqkvT = (_Float16*)(ws + 0);          // [1536][512]  1572864
  _Float16* woutT = (_Float16*)(ws + 1572864);    // [512][512]    524288
  _Float16* w1T   = (_Float16*)(ws + 2097152);    // [1024][512]  1048576
  _Float16* w2T   = (_Float16*)(ws + 3145728);    // [512][1024]  1048576
  _Float16* hA    = (_Float16*)(ws + 4194304);    // [8192][512]  8388608
  _Float16* qkb   = (_Float16*)(ws + 12582912);   // [8192][1024] 16777216
  _Float16* vTb   = (_Float16*)(ws + 29360128);   // [32*64][2048] 8388608
  float*    x2    = (float*)   (ws + 37748736);   // [8192][512] f32 16777216
  _Float16* hB    = (_Float16*)(ws + 54525952);   // [8192][512]  8388608
  _Float16* hff   = (_Float16*)(ws + 62914560);   // [8192][1024] 16777216
  // total 79691776 bytes (~76 MB)

  // weight conversions fp32 [K][N] -> f16 transposed [N][K]
  transpose_cvt<<<dim3(TRIPLE / 32, DIM / 32), 256, 0, stream>>>(w_qkv, wqkvT, DIM, TRIPLE);
  transpose_cvt<<<dim3(DIM / 32, DIM / 32), 256, 0, stream>>>(w_out, woutT, DIM, DIM);
  transpose_cvt<<<dim3(FF / 32, DIM / 32), 256, 0, stream>>>(w1, w1T, DIM, FF);
  transpose_cvt<<<dim3(DIM / 32, FF / 32), 256, 0, stream>>>(w2, w2T, FF, DIM);

  // LN1 -> hA (f16)
  ln_f16<<<TOK / 8, 256, 0, stream>>>(x, ln1_g, ln1_b, hA);

  // QKV projection (split epilogue): Q|K token-major, V transposed
  gemm_wmma<3><<<dim3(TRIPLE / 128, TOK / 256), 256, 0, stream>>>(
      hA, wqkvT, nullptr, nullptr, nullptr, qkb, vTb, TOK, TRIPLE, DIM);

  // flash attention -> reuse hA for attended output (hA dead after QKV GEMM)
  attn_wmma<<<dim3(SEQ / 128, BATCH * HEADS), 256, 0, stream>>>(qkb, vTb, hA);

  // out projection + b_out + residual x -> x2 (f32)
  gemm_wmma<1><<<dim3(DIM / 128, TOK / 256), 256, 0, stream>>>(
      hA, woutT, b_out, x, x2, nullptr, nullptr, TOK, DIM, DIM);

  // LN2 -> hB (f16)
  ln_f16<<<TOK / 8, 256, 0, stream>>>(x2, ln2_g, ln2_b, hB);

  // MLP up: + b1, exact GELU -> hff (f16)
  gemm_wmma<2><<<dim3(FF / 128, TOK / 256), 256, 0, stream>>>(
      hB, w1T, b1, nullptr, nullptr, hff, nullptr, TOK, FF, DIM);

  // MLP down: + b2 + residual x2 -> out (f32)
  gemm_wmma<1><<<dim3(DIM / 128, TOK / 256), 256, 0, stream>>>(
      hff, w2T, b2, x2, out, nullptr, nullptr, TOK, DIM, FF);
}
